// LphaLoss_79989470921234
// MI455X (gfx1250) — matrix operand order repaired and checked by reading, not compiled, not run on hardware
//
#include <hip/hip_runtime.h>
#include <hip/hip_bf16.h>

typedef __attribute__((ext_vector_type(16))) _Float16 v16h;
typedef __attribute__((ext_vector_type(8)))  _Float16 v8h;
typedef __attribute__((ext_vector_type(8)))  float    v8f;

// ---------------------------------------------------------------------------
// Problem geometry: B=16, C=3, H=W=256, 32x32 blocks -> 1024 blocks per tensor.
// Feature blocks are indexed 0..1023 (pred1) and 1024..2047 (target):
//   blk = tensor*1024 + img*64 + by*8 + bx
// Feature layout everywhere: NHWC f16  [blk][h*w][c]
// ---------------------------------------------------------------------------

// ========================= weight packing ==================================
// Pack OIHW f32 weights into WMMA B-fragment order:
//   dst[((kt*NTN + nT)*32 + lane)*16 + e] = w[n][c][dy][dx]
//   n = nT*16 + (lane&15);  k = kt*32 + (e&7) + 8*(lane>>4) + 16*(e>>3);  tap=k/CIN; c=k%CIN
__global__ __launch_bounds__(256) void pack_weights(const float* __restrict__ w,
                                                    _Float16* __restrict__ dst,
                                                    int CIN, int COUT) {
  int total = CIN * 9 * COUT;
  int idx = blockIdx.x * 256 + threadIdx.x;
  if (idx >= total) return;
  int e    = idx & 15;
  int lane = (idx >> 4) & 31;
  int tn   = idx >> 9;
  int NTN  = COUT >> 4;
  int nT   = tn % NTN;
  int kt   = tn / NTN;
  int n = nT * 16 + (lane & 15);
  int k = kt * 32 + (e & 7) + ((lane >> 4) << 3) + ((e >> 3) << 4);
  int tap = k / CIN;
  int c   = k % CIN;
  dst[idx] = (_Float16)w[((size_t)n * CIN + c) * 9 + tap];
}

// ========================= conv1_1 (3->64), VALU ===========================
// Fused: block extraction + ImageNet normalize + 3x3 conv + bias + relu.
__global__ __launch_bounds__(256) void conv1_blocks(
    const float* __restrict__ pred1, const float* __restrict__ target,
    const float* __restrict__ w1, const float* __restrict__ b1,
    _Float16* __restrict__ fout) {
  __shared__ float s_in[32 * 32 * 3];   // [y][x][c], normalized
  __shared__ float s_w[64 * 27];
  __shared__ float s_b[64];
  const float MEANC[3] = {0.485f, 0.456f, 0.406f};
  const float RSTD[3]  = {1.0f / 0.229f, 1.0f / 0.224f, 1.0f / 0.225f};

  int blk = blockIdx.x;
  int tid = threadIdx.x;
  const float* src = (blk < 1024) ? pred1 : target;
  int bb  = blk & 1023;
  int img = bb >> 6;
  int by  = (bb >> 3) & 7;
  int bx  = bb & 7;

  for (int i = tid; i < 32 * 32 * 3; i += 256) {
    int y = i / 96, rr = i % 96, x = rr / 3, c = rr % 3;
    float v = src[(((size_t)img * 3 + c) * 256 + (by * 32 + y)) * 256 + bx * 32 + x];
    s_in[i] = (v - MEANC[c]) * RSTD[c];
  }
  for (int i = tid; i < 64 * 27; i += 256) s_w[i] = w1[i];
  if (tid < 64) s_b[tid] = b1[tid];
  __syncthreads();

  for (int idx = tid; idx < 1024 * 64; idx += 256) {
    int m = idx >> 6, o = idx & 63;
    int y = m >> 5, x = m & 31;
    float acc = s_b[o];
    for (int t = 0; t < 9; ++t) {
      int iy = y + t / 3 - 1, ix = x + t % 3 - 1;
      if (iy >= 0 && iy < 32 && ix >= 0 && ix < 32) {
        const float* pin = &s_in[(iy * 32 + ix) * 3];
        const float* pw  = &s_w[o * 27 + t];
        acc += pin[0] * pw[0] + pin[1] * pw[9] + pin[2] * pw[18];
      }
    }
    fout[((size_t)blk * 1024 + m) * 64 + o] = (_Float16)fmaxf(acc, 0.0f);
  }
}

// ========================= WMMA implicit-GEMM conv =========================
// One workgroup (8 waves) per image block. GEMM: M=HW*HW, K=CIN*9, N=COUT.
// Each wave owns (16-row M tile) x (32-col N pair), loops K in 32-chunks.
template <int HW, int CIN, int COUT, bool RELU, bool OUTF32>
__global__ __launch_bounds__(256) void conv_wmma(
    const _Float16* __restrict__ fin,    // [2048][HW*HW][CIN]
    const _Float16* __restrict__ wp,     // packed B fragments
    const float* __restrict__ bias,      // [COUT]
    _Float16* __restrict__ fout,         // [2048][HW*HW][COUT] (if !OUTF32)
    float* __restrict__ fout32) {        // (if OUTF32)
  constexpr int M   = HW * HW;
  constexpr int NK  = (CIN * 9) / 32;   // K tiles of 32
  constexpr int NTN = COUT / 16;        // N tiles
  constexpr int NTM = M / 16;           // M tiles
  constexpr int NP  = NTN / 2;          // N tile pairs
  constexpr int TOT = NTM * NP;

  int blk  = blockIdx.x;
  int wave = threadIdx.x >> 5;
  int lane = threadIdx.x & 31;
  int lrow  = lane & 15;
  int kHalf = lane >> 4;
  const _Float16* in = fin + (size_t)blk * M * CIN;
  const _Float16 z = (_Float16)0.0f;

  for (int t = wave; t < TOT; t += 8) {
    int mT = t / NP, nP = t % NP;
    int mBase = mT * 16, nT0 = nP * 2;
    int m = mBase + lrow;
    int y = m / HW, x = m % HW;

    v8f acc0 = {};
    v8f acc1 = {};
    for (int kt = 0; kt < NK; ++kt) {
      // ---- A fragment: implicit im2col. One 3x3 tap per 32-wide K tile.
      int tap = (kt * 32) / CIN;
      int cb  = (kt * 32) % CIN;
      int iy = y + tap / 3 - 1;
      int ix = x + tap % 3 - 1;
      bool valid = (iy >= 0) & (iy < HW) & (ix >= 0) & (ix < HW);
      int iyc = valid ? iy : 0;
      int ixc = valid ? ix : 0;
      const _Float16* ap = in + ((size_t)(iyc * HW + ixc)) * CIN + cb + kHalf * 8;
      v8h lo = *(const v8h*)ap;          // K offsets {0..7}+8*kHalf
      v8h hi = *(const v8h*)(ap + 16);   // K offsets {16..23}+8*kHalf
      v16h a;
#pragma unroll
      for (int i = 0; i < 8; ++i) {
        a[i]     = valid ? lo[i] : z;
        a[i + 8] = valid ? hi[i] : z;
      }
      // ---- B fragments: pre-packed, one 32B load each.
      const _Float16* bp0 = wp + ((size_t)(kt * NTN + nT0) * 512) + lane * 16;
      v16h b0 = *(const v16h*)bp0;
      v16h b1 = *(const v16h*)(bp0 + 512);
      if (kt + 1 < NK)  // keep next weight fragment hot
        __builtin_prefetch(wp + ((size_t)((kt + 1) * NTN + nT0) * 512) + lane * 16, 0, 0);

      acc0 = __builtin_amdgcn_wmma_f32_16x16x32_f16(false, a, false, b0,
                                                    (short)0, acc0, false, false);
      acc1 = __builtin_amdgcn_wmma_f32_16x16x32_f16(false, a, false, b1,
                                                    (short)0, acc1, false, false);
    }
    // ---- epilogue: bias (+relu), scatter D per the 16x16 f32 C/D layout.
    int n0 = nT0 * 16 + lrow;
    int n1 = n0 + 16;
    float bi0 = bias[n0], bi1 = bias[n1];
#pragma unroll
    for (int r = 0; r < 8; ++r) {
      int mm = mBase + r + 8 * kHalf;
      float v0 = acc0[r] + bi0;
      float v1 = acc1[r] + bi1;
      if (RELU) { v0 = fmaxf(v0, 0.0f); v1 = fmaxf(v1, 0.0f); }
      size_t orow = ((size_t)blk * M + mm) * COUT;
      if constexpr (OUTF32) {
        fout32[orow + n0] = v0;
        fout32[orow + n1] = v1;
      } else {
        fout[orow + n0] = (_Float16)v0;
        fout[orow + n1] = (_Float16)v1;
      }
    }
  }
}

// ========================= 2x2 max pool (NHWC f16) =========================
__global__ __launch_bounds__(256) void pool2x2(const _Float16* __restrict__ fin,
                                               _Float16* __restrict__ fout,
                                               int HW, int C) {
  int OH = HW >> 1;
  size_t total = (size_t)2048 * OH * OH * C;
  size_t idx = (size_t)blockIdx.x * 256 + threadIdx.x;
  if (idx >= total) return;
  int c = (int)(idx % C);
  size_t r = idx / C;
  int ox = (int)(r % OH); r /= OH;
  int oy = (int)(r % OH);
  int blk = (int)(r / OH);
  size_t ib = (size_t)blk * HW * HW;
  int y0 = oy * 2, x0 = ox * 2;
  float a = (float)fin[(ib + y0 * HW + x0) * C + c];
  float b = (float)fin[(ib + y0 * HW + x0 + 1) * C + c];
  float d = (float)fin[(ib + (y0 + 1) * HW + x0) * C + c];
  float e = (float)fin[(ib + (y0 + 1) * HW + x0 + 1) * C + c];
  fout[idx] = (_Float16)fmaxf(fmaxf(a, b), fmaxf(d, e));
}

// ============ 8x8 FFT2 -> angle -> cosine-sim accumulators =================
// One workgroup per (pred, target) block pair; one thread per channel (256).
// Fully unrolled 2D DFT with constant twiddles; deterministic LDS reduction.
__global__ __launch_bounds__(256) void fft_phase_sim(const float* __restrict__ f3,
                                                     float* __restrict__ sums) {
  int p  = blockIdx.x;       // 0..1023
  int ch = threadIdx.x;      // 0..255
  const float* a = f3 + (size_t)p * 64 * 256 + ch;
  const float* b = f3 + (size_t)(p + 1024) * 64 * 256 + ch;

  float x1[64], x2[64];
#pragma unroll
  for (int s = 0; s < 64; ++s) {
    x1[s] = a[(size_t)s * 256];
    x2[s] = b[(size_t)s * 256];
  }
  const float C8[8] = {1.f,  0.70710678f, 0.f, -0.70710678f,
                       -1.f, -0.70710678f, 0.f, 0.70710678f};
  const float S8[8] = {0.f, -0.70710678f, -1.f, -0.70710678f,
                       0.f,  0.70710678f,  1.f, 0.70710678f};
  float dot = 0.f, n1 = 0.f, n2 = 0.f;
#pragma unroll
  for (int u = 0; u < 8; ++u) {
    float tR1[8], tI1[8], tR2[8], tI2[8];
#pragma unroll
    for (int xx = 0; xx < 8; ++xx) {
      float aR = 0.f, aI = 0.f, bR = 0.f, bI = 0.f;
#pragma unroll
      for (int yy = 0; yy < 8; ++yy) {
        int w = (u * yy) & 7;
        float v1 = x1[yy * 8 + xx], v2 = x2[yy * 8 + xx];
        aR += v1 * C8[w]; aI += v1 * S8[w];
        bR += v2 * C8[w]; bI += v2 * S8[w];
      }
      tR1[xx] = aR; tI1[xx] = aI; tR2[xx] = bR; tI2[xx] = bI;
    }
#pragma unroll
    for (int v = 0; v < 8; ++v) {
      float fR1 = 0.f, fI1 = 0.f, fR2 = 0.f, fI2 = 0.f;
#pragma unroll
      for (int xx = 0; xx < 8; ++xx) {
        int w = (v * xx) & 7;
        float cw = C8[w], sw = S8[w];
        fR1 += tR1[xx] * cw - tI1[xx] * sw;
        fI1 += tR1[xx] * sw + tI1[xx] * cw;
        fR2 += tR2[xx] * cw - tI2[xx] * sw;
        fI2 += tR2[xx] * sw + tI2[xx] * cw;
      }
      float ph1 = atan2f(fI1, fR1);
      float ph2 = atan2f(fI2, fR2);
      dot += ph1 * ph2;
      n1 += ph1 * ph1;
      n2 += ph2 * ph2;
    }
  }
  __shared__ float s_red[3 * 256];
  s_red[ch] = dot; s_red[256 + ch] = n1; s_red[512 + ch] = n2;
  __syncthreads();
  for (int s = 128; s > 0; s >>= 1) {
    if (ch < s) {
      s_red[ch]       += s_red[ch + s];
      s_red[256 + ch] += s_red[256 + ch + s];
      s_red[512 + ch] += s_red[512 + ch + s];
    }
    __syncthreads();
  }
  if (ch == 0) {
    sums[p * 3 + 0] = s_red[0];
    sums[p * 3 + 1] = s_red[256];
    sums[p * 3 + 2] = s_red[512];
  }
}

__global__ __launch_bounds__(256) void make_mask(const float* __restrict__ sums,
                                                 float* __restrict__ mask) {
  int idx = blockIdx.x * 256 + threadIdx.x;
  if (idx >= 1024) return;
  float d  = sums[idx * 3 + 0];
  float q1 = sums[idx * 3 + 1];
  float q2 = sums[idx * 3 + 2];
  float den = fmaxf(sqrtf(q1) * sqrtf(q2), 1e-8f);
  mask[idx] = (d / den >= 0.2f) ? 1.0f : 0.0f;
}

// ========================= masked L1 reductions ============================
__global__ __launch_bounds__(256) void l1_partial(const float* __restrict__ pred2,
                                                  const float* __restrict__ target,
                                                  const float* __restrict__ mask,
                                                  float* __restrict__ partials) {
  __shared__ float sh[256];
  int tid = threadIdx.x;
  float s = 0.f;
#pragma unroll
  for (int j = 0; j < 4; ++j) {
    int idx = blockIdx.x * 1024 + j * 256 + tid;     // < 16*3*256*256
    int bimg = idx / 196608;
    int r    = idx % 196608;
    int pix  = r & 65535;
    int y = pix >> 8, x = pix & 255;
    float m = mask[bimg * 64 + (y >> 5) * 8 + (x >> 5)];
    s += m * fabsf(pred2[idx] - target[idx]);
  }
  sh[tid] = s;
  __syncthreads();
  for (int k = 128; k > 0; k >>= 1) {
    if (tid < k) sh[tid] += sh[tid + k];
    __syncthreads();
  }
  if (tid == 0) partials[blockIdx.x] = sh[0];
}

__global__ __launch_bounds__(256) void finalize(const float* __restrict__ partials,
                                                int nPart,
                                                const float* __restrict__ mask,
                                                float* __restrict__ out) {
  __shared__ float sh[256];
  __shared__ float sh2[256];
  int tid = threadIdx.x;
  float s = 0.f, mc = 0.f;
  for (int i = tid; i < nPart; i += 256) s += partials[i];
  for (int i = tid; i < 1024; i += 256) mc += mask[i];
  sh[tid] = s; sh2[tid] = mc;
  __syncthreads();
  for (int k = 128; k > 0; k >>= 1) {
    if (tid < k) { sh[tid] += sh[tid + k]; sh2[tid] += sh2[tid + k]; }
    __syncthreads();
  }
  if (tid == 0) out[0] = sh[0] / (sh2[0] * 1024.0f + 1e-6f);
}

// ========================= host-side orchestration =========================
extern "C" void kernel_launch(void* const* d_in, const int* in_sizes, int n_in,
                              void* d_out, int out_size, void* d_ws, size_t ws_size,
                              hipStream_t stream) {
  const float* pred1  = (const float*)d_in[0];
  const float* pred2  = (const float*)d_in[1];
  const float* target = (const float*)d_in[2];
  const float* w1 = (const float*)d_in[3];  const float* b1 = (const float*)d_in[4];
  const float* w2 = (const float*)d_in[5];  const float* b2 = (const float*)d_in[6];
  const float* w3 = (const float*)d_in[7];  const float* b3 = (const float*)d_in[8];
  const float* w4 = (const float*)d_in[9];  const float* b4 = (const float*)d_in[10];
  const float* w5 = (const float*)d_in[11]; const float* b5 = (const float*)d_in[12];

  // workspace layout
  char* wsB = (char*)d_ws;
  _Float16* wsH = (_Float16*)d_ws;
  _Float16* W2 = wsH + 0;        //  576* 64 halves
  _Float16* W3 = wsH + 36864;    //  576*128
  _Float16* W4 = wsH + 110592;   // 1152*128
  _Float16* W5 = wsH + 258048;   // 1152*256
  const size_t OFF_FA = 552960ull * 2;               // 1,105,920 B (32B aligned)
  const size_t OFF_FB = OFF_FA + 268435456ull;
  const size_t OFF_SM = OFF_FB + 268435456ull;
  const size_t OFF_MK = OFF_SM + 12288ull;
  const size_t OFF_PT = OFF_MK + 4096ull;
  _Float16* FA = (_Float16*)(wsB + OFF_FA);
  _Float16* FB = (_Float16*)(wsB + OFF_FB);
  float* FAf      = (float*)(wsB + OFF_FA);
  float* sums     = (float*)(wsB + OFF_SM);
  float* mask     = (float*)(wsB + OFF_MK);
  float* partials = (float*)(wsB + OFF_PT);

  // 1) pack weights into WMMA B-fragment layout (L2-resident, ~1 MB)
  pack_weights<<<(576 * 64  + 255) / 256, 256, 0, stream>>>(w2, W2, 64, 64);
  pack_weights<<<(576 * 128 + 255) / 256, 256, 0, stream>>>(w3, W3, 64, 128);
  pack_weights<<<(1152 * 128 + 255) / 256, 256, 0, stream>>>(w4, W4, 128, 128);
  pack_weights<<<(1152 * 256 + 255) / 256, 256, 0, stream>>>(w5, W5, 128, 256);

  // 2) conv1_1 (+norm, +relu), direct VALU: -> FA [2048][1024][64]
  conv1_blocks<<<2048, 256, 0, stream>>>(pred1, target, w1, b1, FA);

  // 3) conv1_2 WMMA: FA -> FB [2048][1024][64]
  conv_wmma<32, 64, 64, true, false><<<2048, 256, 0, stream>>>(FA, W2, b2, FB, nullptr);
  // 4) pool1: FB -> FA [2048][256][64]
  pool2x2<<<(2048 * 256 * 64 + 255) / 256, 256, 0, stream>>>(FB, FA, 32, 64);
  // 5) conv2_1 WMMA: FA -> FB [2048][256][128]
  conv_wmma<16, 64, 128, true, false><<<2048, 256, 0, stream>>>(FA, W3, b3, FB, nullptr);
  // 6) conv2_2 WMMA: FB -> FA [2048][256][128]
  conv_wmma<16, 128, 128, true, false><<<2048, 256, 0, stream>>>(FB, W4, b4, FA, nullptr);
  // 7) pool2: FA -> FB [2048][64][128]
  pool2x2<<<(2048 * 64 * 128 + 255) / 256, 256, 0, stream>>>(FA, FB, 16, 128);
  // 8) conv3_1 WMMA (no relu, f32 out): FB -> FAf [2048][64][256]
  conv_wmma<8, 128, 256, false, true><<<2048, 256, 0, stream>>>(FB, W5, b5, nullptr, FAf);

  // 9) FFT2 phase cosine-sim accumulators per block pair
  fft_phase_sim<<<1024, 256, 0, stream>>>(FAf, sums);
  // 10) threshold mask
  make_mask<<<4, 256, 0, stream>>>(sums, mask);
  // 11) masked L1 partial sums (16*3*256*256 / 1024 = 3072 groups)
  l1_partial<<<3072, 256, 0, stream>>>(pred2, target, mask, partials);
  // 12) final scalar
  finalize<<<1, 256, 0, stream>>>(partials, 3072, mask, (float*)d_out);
}
